// MemTransformerLM_22308060135840
// MI455X (gfx1250) — matrix operand order repaired and checked by reading, not compiled, not run on hardware
//
#include <hip/hip_runtime.h>
#include <hip/hip_bf16.h>
#include <math.h>

// ---------------- model constants ----------------
#define NL    6
#define NH    16
#define DM    1024
#define DH    64
#define DI    4096
#define NTOK  32000
#define BSZ   4
#define QLEN  512
#define MLEN  512
#define KLENT 1024
#define ROWS  (BSZ * QLEN)   // 2048
#define NEGV  (-1.0e30f)
#define VSPLIT 25            // 32000 = 25 * 1280, 1280 = 20 groups of 4 n-tiles

typedef __bf16 bf16;
typedef __attribute__((ext_vector_type(16))) __bf16 bf16x16;
typedef __attribute__((ext_vector_type(8)))  __bf16 bf16x8;
typedef __attribute__((ext_vector_type(8)))  float  f32x8;
typedef __attribute__((ext_vector_type(4)))  unsigned int u32x4;
typedef __attribute__((ext_vector_type(4)))  int i32x4;
typedef __attribute__((ext_vector_type(8)))  int i32x8;

#if defined(__AMDGCN__) && __has_builtin(__builtin_amdgcn_tensor_load_to_lds)
#define USE_TDM 1
#else
#define USE_TDM 0
#endif

// ---------------- WMMA + fragment helpers ----------------
__device__ __forceinline__ f32x8 wmma_bf16(bf16x16 a, bf16x16 b, f32x8 c) {
  return __builtin_amdgcn_wmma_f32_16x16x32_bf16(false, a, false, b, (short)0, c, false, false);
}

// A-matrix 16x32 bf16 fragment: row = row_base + (lane&15);
// lanes 0-15 hold K {0..7,16..23}, lanes 16-31 hold K {8..15,24..31}.
__device__ __forceinline__ bf16x16 frag_a(const bf16* base, int row_base, int stride,
                                          int koff, int lane) {
  const int kb = (lane & 16) >> 1;  // 0 or 8
  const bf16* p = base + (size_t)(row_base + (lane & 15)) * stride + koff + kb;
  union { bf16x16 v; bf16x8 h[2]; } u;
  u.h[0] = *(const bf16x8*)(p);
  u.h[1] = *(const bf16x8*)(p + 16);
  return u.v;
}

// B-matrix 32x16 bf16 fragment: col = col_base + (lane&15);
// lanes 0-15 hold K {0..7,16..23}, lanes 16-31 hold K {8..15,24..31}.
__device__ __forceinline__ bf16x16 frag_b(const bf16* base, int col_base, int stride,
                                          int koff, int lane) {
  const int kb = (lane & 16) >> 1;  // 0 or 8
  const bf16* p = base + (size_t)(col_base + (lane & 15)) * stride + koff + kb;
  union { bf16x16 v; bf16x8 h[2]; } u;
  u.h[0] = *(const bf16x8*)(p);
  u.h[1] = *(const bf16x8*)(p + 16);
  return u.v;
}

#if USE_TDM
// Tensor Data Mover: DMA a 128-row x 64-elt bf16 tile (row stride = K elts) from
// global to LDS at lds_off.  D# pad: 16B pad after every 128B row -> LDS row
// stride 72 elts, matching the bank-conflict-free fragment reader layout.
__device__ __forceinline__ void tdm_load_tile_128x64(const bf16* gsrc, unsigned lds_off,
                                                     int k_elts, int tensor_rows) {
  const unsigned long long ga = (unsigned long long)(size_t)gsrc;
  const unsigned galo = __builtin_amdgcn_readfirstlane((unsigned)(ga & 0xFFFFFFFFull));
  const unsigned gahi = __builtin_amdgcn_readfirstlane((unsigned)((ga >> 32) & 0x01FFFFFFull));
  const unsigned ldsu = __builtin_amdgcn_readfirstlane(lds_off);
  const unsigned ku   = __builtin_amdgcn_readfirstlane((unsigned)k_elts);
  const unsigned ru   = __builtin_amdgcn_readfirstlane((unsigned)tensor_rows);
  u32x4 g0;
  g0[0] = 1u;                       // count=1 valid user descriptor
  g0[1] = ldsu;                     // lds_addr (bytes)
  g0[2] = galo;                     // global_addr[31:0]
  g0[3] = gahi | (2u << 30);        // global_addr[56:32] | type=2 (image)
  i32x8 g1;
  g1[0] = (int)((1u << 16) |        // data_size = 2 bytes
                (1u << 20) |        // pad_enable
                (4u << 22) |        // pad_interval: every 32 DWORDs (128B)
                (3u << 25));        // pad_amount: 4 DWORDs (16B)
  g1[1] = (int)((ku & 0xFFFFu) << 16);                    // tensor_dim0[15:0] @ bits63:48
  g1[2] = (int)(((ku >> 16) & 0xFFFFu) | ((ru & 0xFFFFu) << 16)); // dim0 hi | dim1 lo
  g1[3] = (int)(((ru >> 16) & 0xFFFFu) | (64u << 16));    // dim1 hi | tile_dim0=64
  g1[4] = 128;                                            // tile_dim1=128
  g1[5] = (int)ku;                                        // tensor_dim0_stride[31:0]
  g1[6] = 0;
  g1[7] = 0;
  i32x4 gz = {0, 0, 0, 0};
#if __clang_major__ >= 23
  i32x8 gz8 = {0, 0, 0, 0, 0, 0, 0, 0};
  __builtin_amdgcn_tensor_load_to_lds(g0, g1, gz, gz, gz8, 0);
#else
  __builtin_amdgcn_tensor_load_to_lds(g0, g1, gz, gz, 0);
#endif
}
#endif  // USE_TDM

// ---------------- generic NT bf16 GEMM: C[M,N] = A[M,K] * B[N,K]^T ----------------
// 256 threads = 8 waves, 128x128 block tile, wave = 32x64 (2x4 accs).
// K is staged 64 at a time (16 WMMA/wave per barrier round), via TDM when available.
__global__ __launch_bounds__(256) void gemm_nt_kernel(
    const bf16* __restrict__ A, const bf16* __restrict__ B,
    const float* __restrict__ bias, float* __restrict__ Cf, bf16* __restrict__ Cb,
    int M, int N, int K, int relu) {
  __shared__ __align__(16) bf16 As[128][72];
  __shared__ __align__(16) bf16 Bs[128][72];
  const int tid = threadIdx.x, lane = tid & 31, w = tid >> 5;
  const int wm = w & 3, wn = w >> 2;
  const int nb = N >> 7;
  const int m0 = (blockIdx.x / nb) << 7;
  const int n0 = (blockIdx.x % nb) << 7;
  f32x8 acc[2][4] = {};
  for (int k0 = 0; k0 < K; k0 += 64) {
    __syncthreads();
#if USE_TDM
    if (w == 0) {
      tdm_load_tile_128x64(A + (size_t)m0 * K + k0, (unsigned)(size_t)(void*)&As[0][0], K, M);
      tdm_load_tile_128x64(B + (size_t)n0 * K + k0, (unsigned)(size_t)(void*)&Bs[0][0], K, N);
      __builtin_amdgcn_s_wait_tensorcnt(0);
    }
#else
    {
      const bf16* src = (tid < 128) ? (A + (size_t)(m0 + tid) * K + k0)
                                    : (B + (size_t)(n0 + (tid - 128)) * K + k0);
      bf16* dst = (tid < 128) ? &As[tid][0] : &Bs[tid - 128][0];
#pragma unroll
      for (int u = 0; u < 8; ++u)
        *(u32x4*)(dst + u * 8) = *(const u32x4*)(src + u * 8);
      if (k0 + 64 < K) __builtin_prefetch((const void*)(src + 64), 0, 3);
    }
#endif
    __syncthreads();
#pragma unroll
    for (int ks = 0; ks < 64; ks += 32) {
      bf16x16 af[2], bfr[4];
#pragma unroll
      for (int i = 0; i < 2; ++i) af[i] = frag_a(&As[0][0], wm * 32 + i * 16, 72, ks, lane);
#pragma unroll
      for (int j = 0; j < 4; ++j) bfr[j] = frag_b(&Bs[0][0], wn * 64 + j * 16, 72, ks, lane);
#pragma unroll
      for (int i = 0; i < 2; ++i)
#pragma unroll
        for (int j = 0; j < 4; ++j)
          acc[i][j] = wmma_bf16(af[i], bfr[j], acc[i][j]);
    }
  }
  const int hi8 = (lane & 16) ? 8 : 0, nl = lane & 15;
#pragma unroll
  for (int i = 0; i < 2; ++i)
#pragma unroll
    for (int j = 0; j < 4; ++j)
#pragma unroll
      for (int v = 0; v < 8; ++v) {
        const int m = m0 + wm * 32 + i * 16 + v + hi8;
        const int n = n0 + wn * 64 + j * 16 + nl;
        float val = acc[i][j][v];
        if (bias) val += bias[n];
        if (relu) val = fmaxf(val, 0.0f);
        if (Cf) Cf[(size_t)m * N + n] = val;
        if (Cb) Cb[(size_t)m * N + n] = (bf16)val;
      }
}

// ---------------- flash attention with Transformer-XL relative shift ----------------
// 128 threads = 4 waves; each wave owns 16 query rows for one (b,h).
__global__ __launch_bounds__(128) void flash_attn_kernel(
    const bf16* __restrict__ qbw, const bf16* __restrict__ qbr,
    const bf16* __restrict__ kmat, const bf16* __restrict__ vt,
    const bf16* __restrict__ rk, bf16* __restrict__ attn) {
  __shared__ __align__(16) float Rl[4][16][52];
  __shared__ __align__(16) bf16  Pl[4][16][40];
  const int tid = threadIdx.x, lane = tid & 31, w = tid >> 5;
  int blk = blockIdx.x;
  const int qblk = blk & 7; blk >>= 3;
  const int h = blk & 15;  const int b = blk >> 4;
  const int qi0 = (qblk * 4 + w) * 16;
  const float scale = 0.125f;  // 1/sqrt(64)
  const size_t bh = (size_t)(b * NH + h);
  const bf16* qwv  = qbw + (bh * QLEN + qi0) * DH;
  const bf16* qrv  = qbr + (bh * QLEN + qi0) * DH;
  const bf16* kbas = kmat + bh * KLENT * DH;
  const bf16* vbas = vt + bh * DH * KLENT;
  const bf16* rbas = rk + (size_t)h * KLENT * DH;
  const bf16x16 aw0 = frag_a(qwv, 0, DH, 0, lane), aw1 = frag_a(qwv, 0, DH, 32, lane);
  const bf16x16 ar0 = frag_a(qrv, 0, DH, 0, lane), ar1 = frag_a(qrv, 0, DH, 32, lane);
  f32x8 O[4] = {};
  float mi[8], li[8];
#pragma unroll
  for (int v = 0; v < 8; ++v) { mi[v] = -3.0e38f; li[v] = 0.0f; }
  const int hi8 = (lane & 16) ? 8 : 0, nl = lane & 15;

  for (int j0 = 0; j0 < qi0 + 16 + MLEN; j0 += 32) {
    // AC = (q+bw) . k^T over 32 keys (two 16-col tiles)
    f32x8 S[2] = {};
#pragma unroll
    for (int t = 0; t < 2; ++t) {
      bf16x16 b0 = frag_b(kbas + (size_t)(j0 + 16 * t) * DH, 0, DH, 0,  lane);
      bf16x16 b1 = frag_b(kbas + (size_t)(j0 + 16 * t) * DH, 0, DH, 32, lane);
      S[t] = wmma_bf16(aw0, b0, S[t]);
      S[t] = wmma_bf16(aw1, b1, S[t]);
    }
    // BD raw window covering jr = j0+496-qi0 .. +47  (shift needs 32+15 cols)
    const int jr0 = j0 + 496 - qi0;
#pragma unroll
    for (int t = 0; t < 3; ++t) {
      f32x8 R = {};
      bf16x16 b0 = frag_b(rbas + (size_t)(jr0 + 16 * t) * DH, 0, DH, 0,  lane);
      bf16x16 b1 = frag_b(rbas + (size_t)(jr0 + 16 * t) * DH, 0, DH, 32, lane);
      R = wmma_bf16(ar0, b0, R);
      R = wmma_bf16(ar1, b1, R);
#pragma unroll
      for (int v = 0; v < 8; ++v) Rl[w][v + hi8][16 * t + nl] = R[v];
    }
    // rel_shift: BD[i, j0+t] = raw[i, t + 15 - (i-qi0)], add + scale + causal mask
#pragma unroll
    for (int s = 0; s < 2; ++s)
#pragma unroll
      for (int v = 0; v < 8; ++v) {
        const int r = v + hi8;
        const int t = 16 * s + nl;
        float sv = (S[s][v] + Rl[w][r][t + 15 - r]) * scale;
        if (j0 + t > qi0 + r + MLEN) sv = NEGV;
        S[s][v] = sv;
      }
    // online softmax (row stats replicated over the 16 lanes of each half)
    float al[8];
#pragma unroll
    for (int v = 0; v < 8; ++v) {
      float mx = fmaxf(S[0][v], S[1][v]);
      mx = fmaxf(mx, __shfl_xor(mx, 1));
      mx = fmaxf(mx, __shfl_xor(mx, 2));
      mx = fmaxf(mx, __shfl_xor(mx, 4));
      mx = fmaxf(mx, __shfl_xor(mx, 8));
      const float mnew = fmaxf(mi[v], mx);
      const float p0 = __expf(S[0][v] - mnew);
      const float p1 = __expf(S[1][v] - mnew);
      float rs = p0 + p1;
      rs += __shfl_xor(rs, 1);
      rs += __shfl_xor(rs, 2);
      rs += __shfl_xor(rs, 4);
      rs += __shfl_xor(rs, 8);
      const float alpha = __expf(mi[v] - mnew);
      li[v] = li[v] * alpha + rs;
      mi[v] = mnew;
      al[v] = alpha;
      Pl[w][v + hi8][nl]      = (bf16)p0;
      Pl[w][v + hi8][16 + nl] = (bf16)p1;
    }
#pragma unroll
    for (int nt = 0; nt < 4; ++nt)
#pragma unroll
      for (int v = 0; v < 8; ++v) O[nt][v] *= al[v];
    // O += P @ V  (K dim = 32 keys); V is d-major so B frags are contiguous
    const bf16x16 ap = frag_a(&Pl[w][0][0], 0, 40, 0, lane);
#pragma unroll
    for (int nt = 0; nt < 4; ++nt) {
      bf16x16 bv = frag_b(vbas, nt * 16, KLENT, j0, lane);
      O[nt] = wmma_bf16(ap, bv, O[nt]);
    }
  }
  // epilogue: attn[b, q, h*64+d] bf16
#pragma unroll
  for (int nt = 0; nt < 4; ++nt)
#pragma unroll
    for (int v = 0; v < 8; ++v) {
      const int r = v + hi8;
      const int q = qi0 + r;
      const int d = nt * 16 + nl;
      attn[(size_t)(b * QLEN + q) * DM + h * DH + d] = (bf16)(O[nt][v] / li[v]);
    }
}

// ---------------- streaming logsumexp over tied-embedding logits ----------------
// grid (128 row-tiles, 25 vocab splits); 4 waves; wave takes 5 groups of 4 n-tiles.
__global__ __launch_bounds__(128) void logits_lse_kernel(
    const bf16* __restrict__ X, const bf16* __restrict__ E, float* __restrict__ stats) {
  __shared__ float sm[4][16], sl[4][16];
  const int tid = threadIdx.x, lane = tid & 31, w = tid >> 5;
  const int rt = blockIdx.x, vs = blockIdx.y;
  const int m0 = rt * 16;
  const int hi8 = (lane & 16) ? 8 : 0, nl = lane & 15;
  float mi[8], li[8];
#pragma unroll
  for (int v = 0; v < 8; ++v) { mi[v] = -3.0e38f; li[v] = 0.0f; }
  for (int g = w; g < 20; g += 4) {
    const int n_base = vs * 1280 + g * 64;
    f32x8 S[4] = {};
    for (int ks = 0; ks < 32; ++ks) {
      bf16x16 a = frag_a(X, m0, DM, ks * 32, lane);
#pragma unroll
      for (int t = 0; t < 4; ++t) {
        bf16x16 bb = frag_b(E, n_base + t * 16, DM, ks * 32, lane);
        S[t] = wmma_bf16(a, bb, S[t]);
      }
    }
#pragma unroll
    for (int v = 0; v < 8; ++v) {
      float mx = fmaxf(fmaxf(S[0][v], S[1][v]), fmaxf(S[2][v], S[3][v]));
      mx = fmaxf(mx, __shfl_xor(mx, 1));
      mx = fmaxf(mx, __shfl_xor(mx, 2));
      mx = fmaxf(mx, __shfl_xor(mx, 4));
      mx = fmaxf(mx, __shfl_xor(mx, 8));
      const float mnew = fmaxf(mi[v], mx);
      float rs = __expf(S[0][v] - mnew) + __expf(S[1][v] - mnew) +
                 __expf(S[2][v] - mnew) + __expf(S[3][v] - mnew);
      rs += __shfl_xor(rs, 1);
      rs += __shfl_xor(rs, 2);
      rs += __shfl_xor(rs, 4);
      rs += __shfl_xor(rs, 8);
      li[v] = li[v] * __expf(mi[v] - mnew) + rs;
      mi[v] = mnew;
    }
  }
  if (nl == 0) {
#pragma unroll
    for (int v = 0; v < 8; ++v) { sm[w][v + hi8] = mi[v]; sl[w][v + hi8] = li[v]; }
  }
  __syncthreads();
  if (w == 0 && lane < 16) {
    float M = sm[0][lane];
    for (int ww = 1; ww < 4; ++ww) M = fmaxf(M, sm[ww][lane]);
    float L = 0.0f;
    for (int ww = 0; ww < 4; ++ww) L += __expf(sm[ww][lane] - M) * sl[ww][lane];
    const size_t o = ((size_t)(m0 + lane) * VSPLIT + vs) * 2;
    stats[o] = M; stats[o + 1] = L;
  }
}

// nll[m] = logsumexp(logits[m]) - x[m].emb[target[m]]
__global__ __launch_bounds__(256) void nll_kernel(
    const float* __restrict__ X, const float* __restrict__ emb,
    const int* __restrict__ target, const float* __restrict__ stats,
    float* __restrict__ out) {
  __shared__ float red[256];
  const int m = blockIdx.x, tid = threadIdx.x;
  const int tgt = target[m];
  float d = 0.0f;
  for (int c = tid; c < DM; c += 256)
    d += X[(size_t)m * DM + c] * emb[(size_t)tgt * DM + c];
  red[tid] = d;
  __syncthreads();
  for (int s = 128; s > 0; s >>= 1) { if (tid < s) red[tid] += red[tid + s]; __syncthreads(); }
  if (tid == 0) {
    float M = -3.0e38f;
    for (int vs = 0; vs < VSPLIT; ++vs) M = fmaxf(M, stats[((size_t)m * VSPLIT + vs) * 2]);
    float L = 0.0f;
    for (int vs = 0; vs < VSPLIT; ++vs)
      L += __expf(stats[((size_t)m * VSPLIT + vs) * 2] - M) * stats[((size_t)m * VSPLIT + vs) * 2 + 1];
    out[m] = (M + logf(L)) - red[0];
  }
}

// ---------------- residual + layernorm (writes f32 and bf16 streams) ----------------
__global__ __launch_bounds__(256) void ln_residual_kernel(
    float* __restrict__ x, const float* __restrict__ delta,
    const float* __restrict__ g, const float* __restrict__ b, bf16* __restrict__ xb) {
  __shared__ float r1[256], r2[256];
  const int row = blockIdx.x, tid = threadIdx.x;
  float v[4]; float s = 0.0f, s2 = 0.0f;
#pragma unroll
  for (int i = 0; i < 4; ++i) {
    const int c = i * 256 + tid;
    const float y = x[(size_t)row * DM + c] + delta[(size_t)row * DM + c];
    v[i] = y; s += y; s2 += y * y;
  }
  r1[tid] = s; r2[tid] = s2;
  __syncthreads();
  for (int st = 128; st > 0; st >>= 1) {
    if (tid < st) { r1[tid] += r1[tid + st]; r2[tid] += r2[tid + st]; }
    __syncthreads();
  }
  const float mu = r1[0] * (1.0f / DM);
  const float var = r2[0] * (1.0f / DM) - mu * mu;
  const float rs = rsqrtf(var + 1e-5f);
#pragma unroll
  for (int i = 0; i < 4; ++i) {
    const int c = i * 256 + tid;
    const float o = (v[i] - mu) * rs * g[c] + b[c];
    x[(size_t)row * DM + c] = o;
    xb[(size_t)row * DM + c] = (bf16)o;
  }
}

// ---------------- elementwise / layout kernels ----------------
__global__ void cvt_f32_bf16_kernel(const float* __restrict__ in, bf16* __restrict__ out, size_t n) {
  const size_t i = (size_t)blockIdx.x * 256 + threadIdx.x;
  if (i < n) out[i] = (bf16)in[i];
}

// out[n*K+k] = in[k*N+n]  (weight transpose to NT layout)
__global__ void cvt_transpose_kernel(const float* __restrict__ in, bf16* __restrict__ out,
                                     int K, int N) {
  const size_t i = (size_t)blockIdx.x * 256 + threadIdx.x;
  if (i < (size_t)K * N) {
    const int n = (int)(i / K), k = (int)(i % K);
    out[i] = (bf16)in[(size_t)k * N + n];
  }
}

__global__ void embed_gather_kernel(const int* __restrict__ data, const float* __restrict__ emb,
                                    float* __restrict__ x, bf16* __restrict__ xb) {
  const size_t i = (size_t)blockIdx.x * 256 + threadIdx.x;  // ROWS*DM
  const int row = (int)(i >> 10), c = (int)(i & 1023);
  const float v = emb[(size_t)data[row] * DM + c];
  x[i] = v; xb[i] = (bf16)v;
}

__global__ void pos_emb_kernel(bf16* __restrict__ pos) {
  const size_t i = (size_t)blockIdx.x * 256 + threadIdx.x;  // KLENT*DM
  const int p = (int)(i >> 10), c = (int)(i & 1023);
  const float ps = (float)(KLENT - 1 - p);
  float v;
  if (c < 512) {
    const float f = __expf(-((float)(2 * c) / 1024.0f) * 9.210340372f);
    v = __sinf(ps * f);
  } else {
    const float f = __expf(-((float)(2 * (c - 512)) / 1024.0f) * 9.210340372f);
    v = __cosf(ps * f);
  }
  pos[i] = (bf16)v;
}

__global__ void concat_mem_kernel(const float* __restrict__ mem, const bf16* __restrict__ xb,
                                  bf16* __restrict__ xm) {
  const size_t i = (size_t)blockIdx.x * 256 + threadIdx.x;  // BSZ*KLENT*DM
  const int c = (int)(i & 1023);
  const size_t row = i >> 10;
  const int t = (int)(row & 1023), b = (int)(row >> 10);
  xm[i] = (t < MLEN) ? (bf16)mem[((size_t)(b * MLEN + t)) * DM + c]
                     : xb[((size_t)(b * QLEN + (t - MLEN))) * DM + c];
}

__global__ void qbias_kernel(const float* __restrict__ q, const float* __restrict__ bw,
                             const float* __restrict__ br, bf16* __restrict__ qbw,
                             bf16* __restrict__ qbr) {
  const size_t i = (size_t)blockIdx.x * 256 + threadIdx.x;  // ROWS*DM
  const int c = (int)(i & 1023);
  const size_t row = i >> 10;
  const int h = c >> 6, d = c & 63;
  const int b = (int)(row >> 9), qq = (int)(row & 511);
  const size_t o = ((size_t)(b * NH + h) * QLEN + qq) * DH + d;
  const float qv = q[i];
  qbw[o] = (bf16)(qv + bw[c]);
  qbr[o] = (bf16)(qv + br[c]);
}

__global__ void kv_split_kernel(const float* __restrict__ kv, bf16* __restrict__ k,
                                bf16* __restrict__ vt) {
  const size_t i = (size_t)blockIdx.x * 256 + threadIdx.x;  // BSZ*KLENT*DM
  const int c = (int)(i & 1023);
  const size_t row = i >> 10;
  const int t = (int)(row & 1023), b = (int)(row >> 10);
  const int h = c >> 6, d = c & 63;
  const size_t bh = (size_t)(b * NH + h);
  k[(bh * KLENT + t) * DH + d] = (bf16)kv[row * 2048 + c];
  vt[(bh * DH + d) * KLENT + t] = (bf16)kv[row * 2048 + 1024 + c];
}

__global__ void rk_split_kernel(const float* __restrict__ rk, bf16* __restrict__ rkb) {
  const size_t i = (size_t)blockIdx.x * 256 + threadIdx.x;  // KLENT*DM
  const int c = (int)(i & 1023);
  const int jr = (int)(i >> 10);
  const int h = c >> 6, d = c & 63;
  rkb[((size_t)h * KLENT + jr) * DH + d] = (bf16)rk[i];
}

// ---------------- host orchestration ----------------
extern "C" void kernel_launch(void* const* d_in, const int* in_sizes, int n_in,
                              void* d_out, int out_size, void* d_ws, size_t ws_size,
                              hipStream_t stream) {
  (void)in_sizes; (void)n_in; (void)out_size; (void)ws_size;
  const int*   data   = (const int*)d_in[0];
  const int*   target = (const int*)d_in[1];
  const float* memory = (const float*)d_in[2];
  const float* emb    = (const float*)d_in[3];
  const float* Wq     = (const float*)d_in[4];
  const float* Wkv    = (const float*)d_in[5];
  const float* Wr     = (const float*)d_in[6];
  const float* Wo     = (const float*)d_in[7];
  const float* ffW1   = (const float*)d_in[8];
  const float* ffb1   = (const float*)d_in[9];
  const float* ffW2   = (const float*)d_in[10];
  const float* ffb2   = (const float*)d_in[11];
  const float* ln1_g  = (const float*)d_in[12];
  const float* ln1_b  = (const float*)d_in[13];
  const float* ln2_g  = (const float*)d_in[14];
  const float* ln2_b  = (const float*)d_in[15];
  const float* bias_w = (const float*)d_in[16];
  const float* bias_r = (const float*)d_in[17];

  char* p = (char*)d_ws;
  auto alloc = [&](size_t bytes) -> void* {
    void* r = (void*)p; p += (bytes + 255) & ~(size_t)255; return r;
  };
  bf16*  embT  = (bf16*)alloc((size_t)NTOK * DM * 2);
  bf16*  pos   = (bf16*)alloc((size_t)KLENT * DM * 2);
  float* x     = (float*)alloc((size_t)ROWS * DM * 4);
  bf16*  xb    = (bf16*)alloc((size_t)ROWS * DM * 2);
  bf16*  WqT   = (bf16*)alloc((size_t)DM * DM * 2);
  bf16*  WkvT  = (bf16*)alloc((size_t)2 * DM * DM * 2);
  bf16*  WrT   = (bf16*)alloc((size_t)DM * DM * 2);
  bf16*  WoT   = (bf16*)alloc((size_t)DM * DM * 2);
  bf16*  W1T   = (bf16*)alloc((size_t)DM * DI * 2);
  bf16*  W2T   = (bf16*)alloc((size_t)DI * DM * 2);
  bf16*  xm    = (bf16*)alloc((size_t)BSZ * KLENT * DM * 2);
  float* qf    = (float*)alloc((size_t)ROWS * DM * 4);
  float* kvf   = (float*)alloc((size_t)BSZ * KLENT * 2 * DM * 4);
  float* rkf   = (float*)alloc((size_t)KLENT * DM * 4);
  bf16*  qbw   = (bf16*)alloc((size_t)ROWS * DM * 2);
  bf16*  qbr   = (bf16*)alloc((size_t)ROWS * DM * 2);
  bf16*  kbuf  = (bf16*)alloc((size_t)BSZ * NH * KLENT * DH * 2);
  bf16*  vtb   = (bf16*)alloc((size_t)BSZ * NH * DH * KLENT * 2);
  bf16*  rkb   = (bf16*)alloc((size_t)NH * KLENT * DH * 2 + 65536);  // +pad for shift overread
  bf16*  attnb = (bf16*)alloc((size_t)ROWS * DM * 2);
  float* of    = (float*)alloc((size_t)ROWS * DM * 4);
  bf16*  h1b   = (bf16*)alloc((size_t)ROWS * DI * 2);
  float* h2f   = (float*)alloc((size_t)ROWS * DM * 4);
  float* stats = (float*)alloc((size_t)ROWS * VSPLIT * 2 * 4);

  // embedding table to bf16 (NT layout for logits is emb itself: [vocab, d])
  { const size_t n = (size_t)NTOK * DM;
    cvt_f32_bf16_kernel<<<(unsigned)((n + 255) / 256), 256, 0, stream>>>(emb, embT, n); }
  pos_emb_kernel<<<(KLENT * DM) / 256, 256, 0, stream>>>(pos);
  embed_gather_kernel<<<(ROWS * DM) / 256, 256, 0, stream>>>(data, emb, x, xb);

  for (int l = 0; l < NL; ++l) {
    const float* Wq_l  = Wq  + (size_t)l * DM * DM;
    const float* Wkv_l = Wkv + (size_t)l * DM * 2 * DM;
    const float* Wr_l  = Wr  + (size_t)l * DM * DM;
    const float* Wo_l  = Wo  + (size_t)l * DM * DM;
    const float* W1_l  = ffW1 + (size_t)l * DM * DI;
    const float* W2_l  = ffW2 + (size_t)l * DI * DM;

    cvt_transpose_kernel<<<(DM * DM) / 256, 256, 0, stream>>>(Wq_l, WqT, DM, DM);
    cvt_transpose_kernel<<<(2 * DM * DM) / 256, 256, 0, stream>>>(Wkv_l, WkvT, DM, 2 * DM);
    cvt_transpose_kernel<<<(DM * DM) / 256, 256, 0, stream>>>(Wr_l, WrT, DM, DM);
    cvt_transpose_kernel<<<(DM * DM) / 256, 256, 0, stream>>>(Wo_l, WoT, DM, DM);
    cvt_transpose_kernel<<<(DM * DI) / 256, 256, 0, stream>>>(W1_l, W1T, DM, DI);
    cvt_transpose_kernel<<<(DI * DM) / 256, 256, 0, stream>>>(W2_l, W2T, DI, DM);
    concat_mem_kernel<<<(BSZ * KLENT * DM) / 256, 256, 0, stream>>>(
        memory + (size_t)l * BSZ * MLEN * DM, xb, xm);

    gemm_nt_kernel<<<(ROWS / 128) * (DM / 128), 256, 0, stream>>>(
        xb, WqT, nullptr, qf, nullptr, ROWS, DM, DM, 0);
    gemm_nt_kernel<<<((BSZ * KLENT) / 128) * ((2 * DM) / 128), 256, 0, stream>>>(
        xm, WkvT, nullptr, kvf, nullptr, BSZ * KLENT, 2 * DM, DM, 0);
    gemm_nt_kernel<<<(KLENT / 128) * (DM / 128), 256, 0, stream>>>(
        pos, WrT, nullptr, rkf, nullptr, KLENT, DM, DM, 0);

    qbias_kernel<<<(ROWS * DM) / 256, 256, 0, stream>>>(qf, bias_w, bias_r, qbw, qbr);
    kv_split_kernel<<<(BSZ * KLENT * DM) / 256, 256, 0, stream>>>(kvf, kbuf, vtb);
    rk_split_kernel<<<(KLENT * DM) / 256, 256, 0, stream>>>(rkf, rkb);

    flash_attn_kernel<<<BSZ * NH * 8, 128, 0, stream>>>(qbw, qbr, kbuf, vtb, rkb, attnb);

    gemm_nt_kernel<<<(ROWS / 128) * (DM / 128), 256, 0, stream>>>(
        attnb, WoT, nullptr, of, nullptr, ROWS, DM, DM, 0);
    ln_residual_kernel<<<ROWS, 256, 0, stream>>>(x, of, ln1_g + (size_t)l * DM,
                                                 ln1_b + (size_t)l * DM, xb);
    gemm_nt_kernel<<<(ROWS / 128) * (DI / 128), 256, 0, stream>>>(
        xb, W1T, ffb1 + (size_t)l * DI, nullptr, h1b, ROWS, DI, DM, 1);
    gemm_nt_kernel<<<(ROWS / 128) * (DM / 128), 256, 0, stream>>>(
        h1b, W2T, ffb2 + (size_t)l * DM, h2f, nullptr, ROWS, DM, DI, 0);
    ln_residual_kernel<<<ROWS, 256, 0, stream>>>(x, h2f, ln2_g + (size_t)l * DM,
                                                 ln2_b + (size_t)l * DM, xb);
  }

  dim3 lgrid(ROWS / 16, VSPLIT);
  logits_lse_kernel<<<lgrid, 128, 0, stream>>>(xb, embT, stats);
  nll_kernel<<<ROWS, 256, 0, stream>>>(x, emb, target, stats, (float*)d_out);
}
